// Head_51342039056594
// MI455X (gfx1250) — matrix operand order repaired and checked by reading, not compiled, not run on hardware
//
#include <hip/hip_runtime.h>

typedef _Float16 v16h __attribute__((ext_vector_type(16)));
typedef _Float16 v8h  __attribute__((ext_vector_type(8)));
typedef _Float16 v4h  __attribute__((ext_vector_type(4)));
typedef float    v8f  __attribute__((ext_vector_type(8)));
typedef unsigned int v4u  __attribute__((ext_vector_type(4)));
typedef int          v8i_ __attribute__((ext_vector_type(8)));
typedef int          v4i_ __attribute__((ext_vector_type(4)));

#define D_MODEL 1024
#define NB      8
#define T       2048
#define DK      64

__device__ __forceinline__ v16h cat8(v8h lo, v8h hi) {
  v16h r;
#pragma unroll
  for (int i = 0; i < 8; ++i) { r[i] = lo[i]; r[i + 8] = hi[i]; }
  return r;
}

__device__ __forceinline__ v8f wmma16(v16h a, v16h b, v8f c) {
  // v_wmma_f32_16x16x32_f16: (neg_a, A, neg_b, B, c_mod, C, reuse_a, reuse_b)
  return __builtin_amdgcn_wmma_f32_16x16x32_f16(false, a, false, b, (short)0, c,
                                                false, false);
}

// ---------------------------------------------------------------------------
// TDM: async 2D-tile DMA global -> LDS (f16 elements).
//   tile_d0 = contiguous elements per row, tile_d1 = rows,
//   stride0 = global row stride in elements. Tile written dense to LDS.
// D# layout per CDNA5 ISA ch.8 (group0/group1; groups 2-3 zero for 2D).
// ---------------------------------------------------------------------------
__device__ __forceinline__ void tdm_load_tile_2d(unsigned lds_byte_addr,
                                                 const _Float16* gptr,
                                                 unsigned tile_d0, unsigned tile_d1,
                                                 unsigned stride0) {
  unsigned long long ga = (unsigned long long)(uintptr_t)gptr;
  v4u g0;
  g0[0] = 1u;                                        // count=1 valid descriptor
  g0[1] = lds_byte_addr;                             // LDS dest (bytes)
  g0[2] = (unsigned)ga;                              // global_addr[31:0]
  g0[3] = ((unsigned)(ga >> 32) & 0x1ffffffu) | (2u << 30);  // addr[56:32], type=2
  v8i_ g1;
  g1[0] = (int)(1u << 16);                           // wg_mask=0, data_size=1 (2B)
  g1[1] = (int)((tile_d0 & 0xffffu) << 16);          // tensor_dim0[15:0] @ [63:48]
  g1[2] = (int)((tile_d1 & 0xffffu) << 16);          // dim0 hi=0 | tensor_dim1 lo
  g1[3] = (int)((tile_d0 & 0xffffu) << 16);          // dim1 hi=0 | tile_dim0
  g1[4] = (int)(tile_d1 & 0xffffu);                  // tile_dim1 | tile_dim2=0
  g1[5] = (int)stride0;                              // tensor_dim0_stride[31:0]
  g1[6] = 0;                                         // stride0 hi | stride1 lo
  g1[7] = 0;                                         // stride1 hi
  v4i_ gz4 = {0, 0, 0, 0};
  v8i_ gz8 = {0, 0, 0, 0, 0, 0, 0, 0};
  // 6-arg form (clang-23 / therock headers): (g0, g1, g2, g3, g4, cpol)
  __builtin_amdgcn_tensor_load_to_lds(g0, g1, gz4, gz4, gz8, 0);
}

// ---------------------------------------------------------------------------
// Kernel 0: Wt[m][n][k] = W_m[k][n] as f16; Wq pre-scaled by 1/sqrt(64)=0.125
// ---------------------------------------------------------------------------
__global__ void __launch_bounds__(256) prep_weights_kernel(
    const float* __restrict__ Wq, const float* __restrict__ Wk,
    const float* __restrict__ Wv, _Float16* __restrict__ Wt) {
  int idx = blockIdx.x * 256 + threadIdx.x;    // 0 .. 3*64*1024-1
  int m = idx >> 16;                           // matrix id (uniform per block)
  int r = idx & 65535;
  int n = r >> 10;                             // 0..63
  int k = r & 1023;                            // 0..1023
  const float* W = (m == 0) ? Wq : ((m == 1) ? Wk : Wv);
  float v = W[k * DK + n];
  if (m == 0) v *= 0.125f;                     // fold softmax scale into Q
  Wt[idx] = (_Float16)v;
}

// ---------------------------------------------------------------------------
// Kernel 1: QKV projection. Block = 128 thr (4 waves), 16 rows of x.
//   Q,K -> f16 [B*T][64]; V -> f16 transposed [B][64][T]
// ---------------------------------------------------------------------------
__global__ void __launch_bounds__(128) qkv_proj_kernel(
    const float* __restrict__ x, const _Float16* __restrict__ Wt,
    _Float16* __restrict__ Qh, _Float16* __restrict__ Kh,
    _Float16* __restrict__ Vt) {
  __shared__ _Float16 xs[16 * D_MODEL];        // 32 KB
  const int tid = threadIdx.x;
  const size_t row0 = (size_t)blockIdx.x * 16;

  const float4* xg = (const float4*)(x + row0 * D_MODEL);
#pragma unroll
  for (int i = 0; i < 32; ++i) {
    int e = tid + i * 128;                     // float4 index, 0..4095
    float4 f = xg[e];
    v4h h;
    h[0] = (_Float16)f.x; h[1] = (_Float16)f.y;
    h[2] = (_Float16)f.z; h[3] = (_Float16)f.w;
    *(v4h*)(&xs[e * 4]) = h;
  }
  __syncthreads();

  const int lane = tid & 31;
  const int wave = tid >> 5;
  const int half = lane >> 4;
  const int nn   = lane & 15;
  const int n0   = wave * 16;

  v8f acc0 = {}, acc1 = {}, acc2 = {};
#pragma unroll 4
  for (int kb = 0; kb < 32; ++kb) {            // K = 1024 in steps of 32
    const _Float16* ar = &xs[nn * D_MODEL + kb * 32 + half * 16];
    v16h a = *(const v16h*)ar;
    const _Float16* bq = Wt + (size_t)(n0 + nn) * D_MODEL + kb * 32 + half * 16;
    acc0 = wmma16(a, *(const v16h*)(bq),          acc0);   // Q
    acc1 = wmma16(a, *(const v16h*)(bq + 65536),  acc1);   // K
    acc2 = wmma16(a, *(const v16h*)(bq + 131072), acc2);   // V
  }

#pragma unroll
  for (int r = 0; r < 8; ++r) {
    size_t row = row0 + half * 8 + r;
    Qh[row * DK + n0 + nn] = (_Float16)acc0[r];
    Kh[row * DK + n0 + nn] = (_Float16)acc1[r];
    size_t bb = row >> 11;                     // batch
    size_t tt = row & 2047;                    // time
    Vt[(bb * DK + n0 + nn) * (size_t)T + tt] = (_Float16)acc2[r];
  }
}

// ---------------------------------------------------------------------------
// Kernel 2: flash attention. Block = 128 thr (4 waves); wave owns 16 queries.
// K/V tiles staged once per block via TDM async DMA, double-buffered.
// ---------------------------------------------------------------------------
__global__ void __launch_bounds__(128) attn_kernel(
    const _Float16* __restrict__ Qh, const _Float16* __restrict__ Kh,
    const _Float16* __restrict__ Vt, float* __restrict__ out) {
  __shared__ _Float16 ktile[2][32 * DK];       // [key 0..31][d 0..63], 4KB each
  __shared__ _Float16 vtile[2][DK * 32];       // [v 0..63][key 0..31], 4KB each
  __shared__ _Float16 pstage[4][16 * 32];      // per-wave P staging (C->A layout)

  const int tid  = threadIdx.x;
  const int lane = tid & 31;
  const int wave = tid >> 5;
  const int half = lane >> 4;
  const int nn   = lane & 15;

  const int blk = blockIdx.x;                  // 256 blocks
  const int b   = blk >> 5;                    // batch
  const int m   = blk & 31;                    // q-block within batch
  const int q0  = m * 64 + wave * 16;          // first query row of this wave

  const _Float16* Qb = Qh + (size_t)b * T * DK;
  const _Float16* Kb = Kh + (size_t)b * T * DK;
  const _Float16* Vb = Vt + (size_t)b * DK * T;

  // Q A-fragments for kbase = 0 and 32 (d = 64 contraction, 2 WMMA k-steps)
  v16h qa0, qa1;
  {
    const _Float16* p = Qb + (size_t)(q0 + nn) * DK + half * 16;
    qa0 = *(const v16h*)(p);
    qa1 = *(const v16h*)(p + 32);
  }

  float mrow[8], lrow[8];
#pragma unroll
  for (int r = 0; r < 8; ++r) { mrow[r] = -1e30f; lrow[r] = 0.0f; }
  v8f o0 = {}, o1 = {}, o2 = {}, o3 = {};

  _Float16* myp = &pstage[wave][0];
  const int nkt_wave = (q0 + 47) >> 5;         // causal 32-key tiles for this wave
  const int nkt_blk  = 2 * m + 2;              // max over the 4 waves of the block

  unsigned ldsk[2] = { (unsigned)(uintptr_t)&ktile[0][0],
                       (unsigned)(uintptr_t)&ktile[1][0] };
  unsigned ldsv[2] = { (unsigned)(uintptr_t)&vtile[0][0],
                       (unsigned)(uintptr_t)&vtile[1][0] };

  // Prologue: wave 0 kicks off the DMA for tile 0.
  if (wave == 0) {
    tdm_load_tile_2d(ldsk[0], Kb, /*d0=*/DK, /*d1=*/32, /*stride=*/DK);
    tdm_load_tile_2d(ldsv[0], Vb, /*d0=*/32, /*d1=*/DK, /*stride=*/T);
  }

  for (int j = 0; j < nkt_blk; ++j) {
    if (wave == 0) {
      if (j + 1 < nkt_blk) {                   // prefetch tile j+1, then wait tile j
        const int kn = (j + 1) * 32;
        tdm_load_tile_2d(ldsk[(j + 1) & 1], Kb + (size_t)kn * DK, DK, 32, DK);
        tdm_load_tile_2d(ldsv[(j + 1) & 1], Vb + kn, 32, DK, T);
        __builtin_amdgcn_s_wait_tensorcnt(2);  // 2 newest may remain in flight
      } else {
        __builtin_amdgcn_s_wait_tensorcnt(0);
      }
    }
    __syncthreads();                           // tile j visible to all waves

    if (j < nkt_wave) {                        // wave-uniform: EXEC stays all-1s
      const int kt0 = j * 32;
      const _Float16* kt_ = &ktile[j & 1][0];
      const _Float16* vt_ = &vtile[j & 1][0];

      // ---- S = Q @ K^T for 32 keys (two 16-key column tiles) ----
      v8f s0 = {}, s1 = {};
      {
        const _Float16* k0 = kt_ + (size_t)nn * DK + half * 16;
        const _Float16* k1 = kt_ + (size_t)(16 + nn) * DK + half * 16;
        s0 = wmma16(qa0, *(const v16h*)(k0),      s0);
        s0 = wmma16(qa1, *(const v16h*)(k0 + 32), s0);
        s1 = wmma16(qa0, *(const v16h*)(k1),      s1);
        s1 = wmma16(qa1, *(const v16h*)(k1 + 32), s1);
      }

      // ---- causal mask: only the boundary tile crosses the diagonal ----
      if (j == nkt_wave - 1) {
#pragma unroll
        for (int r = 0; r < 8; ++r) {
          int q = q0 + half * 8 + r;
          s0[r] = (kt0 + nn > q)      ? -1e30f : s0[r];
          s1[r] = (kt0 + 16 + nn > q) ? -1e30f : s1[r];
        }
      }

      // ---- online softmax; row lives in one VGPR across a 16-lane group ----
#pragma unroll
      for (int r = 0; r < 8; ++r) {
        float mt = fmaxf(s0[r], s1[r]);
        mt = fmaxf(mt, __shfl_xor(mt, 1, 32));
        mt = fmaxf(mt, __shfl_xor(mt, 2, 32));
        mt = fmaxf(mt, __shfl_xor(mt, 4, 32));
        mt = fmaxf(mt, __shfl_xor(mt, 8, 32));
        float mnew = fmaxf(mrow[r], mt);
        float corr = __expf(mrow[r] - mnew);
        mrow[r] = mnew;
        float p0 = __expf(s0[r] - mnew);
        float p1 = __expf(s1[r] - mnew);
        float ps = p0 + p1;
        ps += __shfl_xor(ps, 1, 32);
        ps += __shfl_xor(ps, 2, 32);
        ps += __shfl_xor(ps, 4, 32);
        ps += __shfl_xor(ps, 8, 32);
        lrow[r] = lrow[r] * corr + ps;
        o0[r] *= corr; o1[r] *= corr; o2[r] *= corr; o3[r] *= corr;
        myp[(half * 8 + r) * 32 + nn]      = (_Float16)p0;
        myp[(half * 8 + r) * 32 + 16 + nn] = (_Float16)p1;
      }

      // ---- O += P @ V : P as A-fragment; V tile rows contiguous in key ----
      const _Float16* pr = myp + nn * 32 + half * 16;
      v16h pa = *(const v16h*)pr;
      const _Float16* vp = vt_ + (size_t)nn * 32 + half * 16;
      o0 = wmma16(pa, *(const v16h*)(vp),           o0);
      o1 = wmma16(pa, *(const v16h*)(vp + 16 * 32), o1);
      o2 = wmma16(pa, *(const v16h*)(vp + 32 * 32), o2);
      o3 = wmma16(pa, *(const v16h*)(vp + 48 * 32), o3);
    }

    __syncthreads();                           // buffer reuse fence (tile j+2)
  }

  // ---- epilogue: normalize and store fp32 ----
#pragma unroll
  for (int r = 0; r < 8; ++r) {
    float inv = 1.0f / lrow[r];
    size_t row = (size_t)b * T + q0 + half * 8 + r;
    out[row * DK + nn]      = o0[r] * inv;
    out[row * DK + 16 + nn] = o1[r] * inv;
    out[row * DK + 32 + nn] = o2[r] * inv;
    out[row * DK + 48 + nn] = o3[r] * inv;
  }
}

// ---------------------------------------------------------------------------
extern "C" void kernel_launch(void* const* d_in, const int* in_sizes, int n_in,
                              void* d_out, int out_size, void* d_ws, size_t ws_size,
                              hipStream_t stream) {
  const float* x  = (const float*)d_in[0];
  const float* Wq = (const float*)d_in[1];
  const float* Wk = (const float*)d_in[2];
  const float* Wv = (const float*)d_in[3];
  float* out = (float*)d_out;

  char* ws = (char*)d_ws;
  const size_t WT_BYTES  = (size_t)3 * DK * D_MODEL * sizeof(_Float16); // 384 KB
  const size_t QKV_BYTES = (size_t)NB * T * DK * sizeof(_Float16);      // 2 MB each
  _Float16* Wt = (_Float16*)(ws);
  _Float16* Qh = (_Float16*)(ws + WT_BYTES);
  _Float16* Kh = (_Float16*)(ws + WT_BYTES + QKV_BYTES);
  _Float16* Vt = (_Float16*)(ws + WT_BYTES + 2 * QKV_BYTES);

  prep_weights_kernel<<<768, 256, 0, stream>>>(Wq, Wk, Wv, Wt);
  qkv_proj_kernel<<<(NB * T) / 16, 128, 0, stream>>>(x, Wt, Qh, Kh, Vt);
  attn_kernel<<<(NB * T) / 64, 128, 0, stream>>>(Qh, Kh, Vt, out);
}
